// mBP_model_41626823033493
// MI455X (gfx1250) — matrix operand order repaired and checked by reading, not compiled, not run on hardware
//
#include <hip/hip_runtime.h>

#define RCUT_   5.0f
#define RCANG_  3.5f
#define MINRC_  0.5f
#define NATOM   512
#define BATCH   4
#define PI_F    3.14159265358979f

typedef __attribute__((ext_vector_type(16))) _Float16 v16h;
typedef __attribute__((ext_vector_type(8)))  float    v8f;

// lxlylz enumeration for zeta=2 (l=0..2), matching reference order
__device__ __constant__ int   LXc[10] = {0,0,0,1,0,0,0,1,1,2};
__device__ __constant__ int   LYc[10] = {0,0,1,0,0,1,2,0,1,0};
__device__ __constant__ int   LZc[10] = {0,1,0,0,2,1,0,1,0,0};
__device__ __constant__ float CPLUSc[10] = {1.f, 2.f, 2.f, 2.f, 1.f, 2.f, 1.f, 2.f, 2.f, 1.f};
__device__ __constant__ float CMINc [10] = {1.f,-2.f,-2.f,-2.f, 1.f, 2.f, 1.f, 2.f, 2.f, 1.f};

__device__ __forceinline__ float powi2(float x, int n) {
    return (n == 0) ? 1.0f : ((n == 1) ? x : x * x);
}

// branch-free tanh on v_exp_f32 (no EXEC-mask libm slow path)
__device__ __forceinline__ float fast_tanh(float x) {
    float ax = fabsf(x);
    float e  = __expf(-2.0f * ax);
    float r  = (1.0f - e) / (1.0f + e);
    return copysignf(r, x);
}

// ---------------- hyper-parameter prep + weight packing into WMMA B-layout ----------
// hyper layout (floats): [0..23] width_rad, [24..47] rs_rad, [48..55] width_ang,
// [56..63] rs_ang, [64..143] wlp[t*10+p] (=0.5*c_plus*w), [144..223] wlm
//
// Packed B layout (per GEMM): halves indexed by
//   (((ks*4 + nt)*32 + lane)*16 + 2*v + piece)
// holding W[32*ks + 2*v + 16*(lane>=16) + piece][nt*16 + (lane&15)]
// so each lane reads its 16 halves as two contiguous uint4 (b128) per k-step.
__global__ void prep_kernel(const float* wrW, const float* wrB,
                            const float* waW, const float* waB,
                            const float* rsrW, const float* rsrB,
                            const float* rsaW, const float* rsaB,
                            const float* wlW, const float* wlB,
                            const float* aw0, const float* aw1,
                            float* hyper, _Float16* w0p, _Float16* w1p) {
    int tid = threadIdx.x;   // 128 threads
    if (tid < 24) {
        float x = wrW[tid] + wrB[tid];
        hyper[tid] = logf(1.0f + __expf(x));                       // softplus
        float y = rsrW[tid] + rsrB[tid];
        hyper[24 + tid] = MINRC_ + (RCUT_ - MINRC_) / (1.0f + __expf(-y));
    }
    if (tid < 8) {
        float x = waW[tid] + waB[tid];
        hyper[48 + tid] = logf(1.0f + __expf(x));
        float y = rsaW[tid] + rsaB[tid];
        hyper[56 + tid] = MINRC_ + (RCANG_ - MINRC_) / (1.0f + __expf(-y));
    }
    if (tid < 80) {
        int p = tid % 10;
        float w = 1.0f / (1.0f + __expf(-(wlW[tid] + wlB[tid])));  // sigmoid
        hyper[64  + tid] = 0.5f * CPLUSc[p] * w;                   // pref=2^(1-zeta)=0.5
        hyper[144 + tid] = 0.5f * CMINc[p]  * w;
    }
    // GEMM1 B: K padded to 160 (5 ksteps), rows >=152 are zero
    for (int idx = tid; idx < 5 * 4 * 32 * 16; idx += 128) {
        int h    = idx & 15;
        int lane = (idx >> 4) & 31;
        int nt   = (idx >> 9) & 3;
        int ks   = idx >> 11;
        int v = h >> 1, piece = h & 1;
        int kb  = 32 * ks + 2 * v + ((lane >= 16) ? 16 : 0) + piece;
        int col = nt * 16 + (lane & 15);
        w0p[idx] = (kb < 152) ? (_Float16)aw0[kb * 64 + col] : (_Float16)0.0f;
    }
    // GEMM2 B: K = 64 (2 ksteps)
    for (int idx = tid; idx < 2 * 4 * 32 * 16; idx += 128) {
        int h    = idx & 15;
        int lane = (idx >> 4) & 31;
        int nt   = (idx >> 9) & 3;
        int ks   = idx >> 11;
        int v = h >> 1, piece = h & 1;
        int kb  = 32 * ks + 2 * v + ((lane >= 16) ? 16 : 0) + piece;
        int col = nt * 16 + (lane & 15);
        w1p[idx] = (_Float16)aw1[kb * 64 + col];
    }
}

// ---------------- per-atom species scalar e[b,n] ----------------
__global__ void species_kernel(const int* species, const float* sw0, const float* sb0,
                               const float* sw1, const float* sb1, float* eArr) {
    int atom = blockIdx.x * blockDim.x + threadIdx.x;
    if (atom >= BATCH * NATOM) return;
    int z = species[atom];
    const float* row = sw0 + (size_t)z * 64;
    float acc = sb1[0];
#pragma unroll 8
    for (int k = 0; k < 64; ++k)
        acc += fast_tanh(row[k] + sb0[k]) * sw1[k];
    eArr[atom] = acc;
}

// ---------------- pairwise descriptor: one block (128 thr) per atom ----------------
// featH layout per atom (f16, ld=160): [0..23] G_rad, [24..87] ang_p (s*8+t),
// [88..151] ang_m, [152..159] zero pad
__global__ void desc_kernel(const float* __restrict__ pos, const float* __restrict__ eArr,
                            const float* __restrict__ hyper, _Float16* __restrict__ featH) {
    const int atom = blockIdx.x;
    const int b = atom >> 9;
    const int i = atom & 511;
    const int tid = threadIdx.x;

    __shared__ float sD[128], sFr[128], sFa[128], sRx[128], sRy[128], sRz[128];
    __shared__ float sM[80];

    const float xi = pos[atom * 3 + 0];
    const float yi = pos[atom * 3 + 1];
    const float zi = pos[atom * 3 + 2];

    // per-thread channel parameters
    float wch = 0.f, rsch = 0.f;
    int lx = 0, ly = 0, lz = 0;
    if (tid < 24) {
        wch = hyper[tid]; rsch = hyper[24 + tid];
    } else if (tid < 104) {
        int c = tid - 24, s = c / 10, p = c % 10;
        wch = hyper[48 + s]; rsch = hyper[56 + s];
        lx = LXc[p]; ly = LYc[p]; lz = LZc[p];
    }

    float acc = 0.0f;
    for (int tile = 0; tile < 4; ++tile) {
        int j = tile * 128 + tid;
        int gj = b * 512 + j;
        float dx = xi - pos[gj * 3 + 0];
        float dy = yi - pos[gj * 3 + 1];
        float dz = zi - pos[gj * 3 + 2];
        float d = sqrtf(dx * dx + dy * dy + dz * dz + 1e-12f);
        float ej = eArr[gj];
        bool mask = (j != i);
        float fr = (mask && d < RCUT_)  ? 0.5f * (__cosf(d * (PI_F / RCUT_))  + 1.0f) : 0.0f;
        float fa = (mask && d < RCANG_) ? 0.5f * (__cosf(d * (PI_F / RCANG_)) + 1.0f) : 0.0f;
        float inv = (d > 1e-6f) ? 1.0f / d : 1.0f;
        sD[tid] = d; sFr[tid] = fr * ej; sFa[tid] = fa * ej;
        sRx[tid] = dx * inv; sRy[tid] = dy * inv; sRz[tid] = dz * inv;
        __syncthreads();

        if (tid < 24) {
            for (int q = 0; q < 128; ++q) {
                float dd = sD[q] - rsch;
                acc += __expf(-wch * dd * dd) * sFr[q];
            }
        } else if (tid < 104) {
            for (int q = 0; q < 128; ++q) {
                float dd = sD[q] - rsch;
                float g = __expf(-wch * dd * dd) * sFa[q];
                float mono = powi2(sRx[q], lx) * powi2(sRy[q], ly) * powi2(sRz[q], lz);
                acc += g * mono;
            }
        }
        __syncthreads();
    }

    _Float16* f = featH + (size_t)atom * 160;
    if (tid >= 24 && tid < 104) sM[tid - 24] = acc;
    if (tid < 24) f[tid] = (_Float16)acc;
    if (tid >= 104 && tid < 112) f[tid + 48] = (_Float16)0.0f;   // pad 152..159
    __syncthreads();

    if (tid < 64) {                                              // tid = s*8 + t
        int s = tid >> 3, t = tid & 7;
        const float* wlp = hyper + 64;
        const float* wlm = hyper + 144;
        float ap = 0.f, am = 0.f;
#pragma unroll
        for (int p = 0; p < 10; ++p) {
            float m = sM[s * 10 + p];
            float m2 = m * m;
            ap += m2 * wlp[t * 10 + p];
            am += m2 * wlm[t * 10 + p];
        }
        f[24 + tid] = (_Float16)ap;
        f[88 + tid] = (_Float16)am;
    }
}

// ---------------- WMMA GEMM + bias + tanh: Out[2048,64] = tanh(X[2048,K]@W[K,64]+b) ----
// One wave per 16x16 tile; KSTEPS fully unrolled; A/B each loaded as 2x b128 per step.
template <int KSTEPS>
__global__ void mlp_gemm_tanh(const _Float16* __restrict__ X,
                              const uint4* __restrict__ Bp,
                              const float* __restrict__ bias,
                              _Float16* __restrict__ Out) {
    int wid  = (blockIdx.x * blockDim.x + threadIdx.x) >> 5;
    int lane = threadIdx.x & 31;
    int mt = wid >> 2, nt = wid & 3;        // 128 row tiles x 4 col tiles
    int m0 = mt * 16, n0 = nt * 16;
    int hi = lane >> 4;
    int lr = lane & 15;
    const _Float16* xrow = X + (size_t)(m0 + lr) * (KSTEPS * 32);  // A: row per lane
    int col = n0 + lr;                                             // B/D: col per lane

    union { v16h v; uint4 q[2]; } A, Bm;
    union { v8f  v; float  f[8]; } C;
#pragma unroll
    for (int t = 0; t < 8; ++t) C.f[t] = 0.0f;

#pragma unroll
    for (int ks = 0; ks < KSTEPS; ++ks) {
        int k0 = ks * 32;
        // A 16x32 f16: lane holds K = k0+16*hi+[0..15] (contiguous)
        A.q[0] = *(const uint4*)(xrow + k0 + hi * 16);
        A.q[1] = *(const uint4*)(xrow + k0 + hi * 16 + 8);
        // B pre-packed per lane: 2 contiguous uint4
        const uint4* bp = Bp + (size_t)((ks * 4 + nt) * 32 + lane) * 2;
        Bm.q[0] = bp[0];
        Bm.q[1] = bp[1];
        C.v = __builtin_amdgcn_wmma_f32_16x16x32_f16(
            false, A.v, false, Bm.v, (short)0, C.v, false, false);
    }

    float bcol = bias[col];
#pragma unroll
    for (int t = 0; t < 8; ++t) {
        int orow = m0 + t + (hi ? 8 : 0);   // D: VGPR t -> row m0+t (+8 for hi half-wave)
        Out[(size_t)orow * 64 + col] = (_Float16)fast_tanh(C.f[t] + bcol);
    }
}

// ---------------- final layer + per-structure sum ----------------
__global__ void energy_kernel(const _Float16* __restrict__ H2, const float* __restrict__ aw2,
                              const float* __restrict__ ab2, float* __restrict__ out) {
    int bidx = blockIdx.x;
    int tid = threadIdx.x;
    __shared__ float red[256];
    float acc = 0.0f;
    for (int i = tid; i < NATOM; i += 256) {
        const _Float16* h = H2 + (size_t)(bidx * NATOM + i) * 64;
        float s = ab2[0];
#pragma unroll 8
        for (int k = 0; k < 64; ++k) s += (float)h[k] * aw2[k];
        acc += s;
    }
    red[tid] = acc;
    __syncthreads();
    for (int off = 128; off > 0; off >>= 1) {
        if (tid < off) red[tid] += red[tid + off];
        __syncthreads();
    }
    if (tid == 0) out[bidx] = red[0];
}

extern "C" void kernel_launch(void* const* d_in, const int* in_sizes, int n_in,
                              void* d_out, int out_size, void* d_ws, size_t ws_size,
                              hipStream_t stream) {
    (void)in_sizes; (void)n_in; (void)out_size; (void)ws_size;
    const float* positions = (const float*)d_in[0];
    const int*   species   = (const int*)  d_in[1];
    const float* aw0 = (const float*)d_in[2];
    const float* ab0 = (const float*)d_in[3];
    const float* aw1 = (const float*)d_in[4];
    const float* ab1 = (const float*)d_in[5];
    const float* aw2 = (const float*)d_in[6];
    const float* ab2 = (const float*)d_in[7];
    const float* sw0 = (const float*)d_in[8];
    const float* sb0 = (const float*)d_in[9];
    const float* sw1 = (const float*)d_in[10];
    const float* sb1 = (const float*)d_in[11];
    const float* wrW = (const float*)d_in[12];
    const float* wrB = (const float*)d_in[13];
    const float* waW = (const float*)d_in[14];
    const float* waB = (const float*)d_in[15];
    const float* rsrW = (const float*)d_in[16];
    const float* rsrB = (const float*)d_in[17];
    const float* rsaW = (const float*)d_in[18];
    const float* rsaB = (const float*)d_in[19];
    const float* wlW = (const float*)d_in[20];
    const float* wlB = (const float*)d_in[21];

    char* ws = (char*)d_ws;
    float*    hyper = (float*)   (ws + 0);        // 224 floats (pad to 1 KiB)
    float*    eArr  = (float*)   (ws + 1024);     // 2048 floats
    _Float16* featH = (_Float16*)(ws + 9216);     // 2048*160 f16 = 655360 B
    _Float16* w0p   = (_Float16*)(ws + 664576);   // packed GEMM1 B: 10240 f16 = 20480 B
    _Float16* w1p   = (_Float16*)(ws + 685056);   // packed GEMM2 B: 4096 f16 = 8192 B
    _Float16* h1    = (_Float16*)(ws + 693248);   // 2048*64 f16
    _Float16* h2    = (_Float16*)(ws + 955392);   // 2048*64 f16

    prep_kernel<<<1, 128, 0, stream>>>(wrW, wrB, waW, waB, rsrW, rsrB, rsaW, rsaB,
                                       wlW, wlB, aw0, aw1, hyper, w0p, w1p);
    species_kernel<<<8, 256, 0, stream>>>(species, sw0, sb0, sw1, sb1, eArr);
    desc_kernel<<<BATCH * NATOM, 128, 0, stream>>>(positions, eArr, hyper, featH);
    mlp_gemm_tanh<5><<<64, 256, 0, stream>>>(featH, (const uint4*)w0p, ab0, h1);
    mlp_gemm_tanh<2><<<64, 256, 0, stream>>>(h1, (const uint4*)w1p, ab1, h2);
    energy_kernel<<<BATCH, 256, 0, stream>>>(h2, aw2, ab2, (float*)d_out);
}
